// TorchMockAttention_13812614824661
// MI455X (gfx1250) — matrix operand order repaired and checked by reading, not compiled, not run on hardware
//
#include <hip/hip_runtime.h>

// ---------------------------------------------------------------------------
// Types
// ---------------------------------------------------------------------------
typedef __bf16 bf16;
typedef __attribute__((ext_vector_type(16))) __bf16 v16bf;
typedef __attribute__((ext_vector_type(8)))  __bf16 v8bf;
typedef __attribute__((ext_vector_type(8)))  float  v8f;
typedef __attribute__((ext_vector_type(4)))  unsigned u32x4;
typedef __attribute__((ext_vector_type(8)))  unsigned u32x8;

#define CAT16(lo, hi) __builtin_shufflevector((lo), (hi), 0,1,2,3,4,5,6,7,8,9,10,11,12,13,14,15)

static __device__ __forceinline__ bf16 f2bf(float f) {
  unsigned u = __builtin_bit_cast(unsigned, f);
  u += 0x7FFFu + ((u >> 16) & 1u);                 // round-to-nearest-even
  unsigned short h = (unsigned short)(u >> 16);
  return __builtin_bit_cast(bf16, h);
}

static __device__ __forceinline__ unsigned pack2bf(float a, float b) {
  unsigned lo = (unsigned)__builtin_bit_cast(unsigned short, f2bf(a));
  unsigned hi = (unsigned)__builtin_bit_cast(unsigned short, f2bf(b));
  return lo | (hi << 16);
}

// ---------------------------------------------------------------------------
// TDM: async-load a 128-row x 32-col bf16 tile (row-major, row stride
// `stride0` elems) from global into LDS at byte offset `lds_addr`, padding
// the LDS destination by 4 DWORDs (8 bf16) after every 16 DWORDs (32 bf16)
// -> matches the [128][40] padded LDS tile layout.  2D descriptor: groups
// 0 and 1 only (VADDR2/3 omitted).  Tracked by TENSORcnt.
// ---------------------------------------------------------------------------
static __device__ __forceinline__ void tdm_load_tile_bf16(
    unsigned lds_addr, const bf16* gsrc, unsigned tdim0, unsigned tdim1,
    unsigned stride0) {
  unsigned long long ga = (unsigned long long)gsrc;
  u32x4 g0 = {
      1u,                                   // count=1: one valid descriptor
      lds_addr,                             // D#.lds_addr
      (unsigned)ga,                         // global_addr[31:0]
      (unsigned)(ga >> 32) | (2u << 30)     // global_addr[56:32] | type=2
  };
  u32x8 g1 = {
      (1u << 16) | (1u << 20) | (3u << 22) | (3u << 25),
      //  data_size=2B | pad_enable | pad_interval=16DW | pad_amount=4DW
      (tdim0 & 0xFFFFu) << 16,                        // tensor_dim0[15:0]
      (tdim0 >> 16) | ((tdim1 & 0xFFFFu) << 16),      // dim0[31:16]|dim1[15:0]
      (tdim1 >> 16) | (32u << 16),                    // dim1[31:16]|tile_dim0=32
      128u,                                           // tile_dim1=128
      stride0,                                        // tensor_dim0_stride[31:0]
      0u, 0u
  };
  asm volatile("tensor_load_to_lds %0, %1" : : "s"(g0), "s"(g1) : "memory");
}

// ---------------------------------------------------------------------------
// fp32 -> bf16 conversion, 4 elements / thread
// ---------------------------------------------------------------------------
__global__ void cvt_f32_to_bf16(const float* __restrict__ in,
                                bf16* __restrict__ out, int n4) {
  int i = blockIdx.x * blockDim.x + threadIdx.x;
  if (i >= n4) return;
  float4 v = ((const float4*)in)[i];
  uint2 p;
  p.x = pack2bf(v.x, v.y);
  p.y = pack2bf(v.z, v.w);
  ((uint2*)out)[i] = p;
}

// ---------------------------------------------------------------------------
// LDS fragment loads per ISA 16-bit WMMA A/B lane layouts (two 16B reads)
// ---------------------------------------------------------------------------
static __device__ __forceinline__ v16bf ld_a_frag(const bf16* rp, int half) {
  v8bf lo = *(const v8bf*)(rp + half * 8);        // K = g*8 .. g*8+7
  v8bf hi = *(const v8bf*)(rp + 16 + half * 8);   // K = 16+g*8 ..
  return CAT16(lo, hi);
}
static __device__ __forceinline__ v16bf ld_b_frag(const bf16* rp, int half) {
  v8bf lo = *(const v8bf*)(rp + half * 8);        // K = g*8 .. g*8+7
  v8bf hi = *(const v8bf*)(rp + 16 + half * 8);   // K = 16+g*8 ..
  return CAT16(lo, hi);
}

// ---------------------------------------------------------------------------
// Tiled bf16 WMMA GEMM:  C[M,N] = alpha * A[M,K] x op(B) (+ mask)
//   BT=true : B is [N,K] row-major (NT).  A and B tiles staged by the Tensor
//             Data Mover, double-buffered (DMA of tile k+1 overlaps WMMAs of
//             tile k), synchronized with s_wait_tensorcnt + barriers.
//   BT=false: B is [K,N] row-major (NN); manual staging w/ LDS transpose.
// Block tile 128x128, K-chunk 32. 256 threads = 8 waves; wave tile 64x32
// (4x2 wmma 16x16 accumulators). blockIdx.z = batch (strides sA/sB/sC elems).
// ---------------------------------------------------------------------------
template <bool BT, bool OUT_BF16, bool ADD_MASK>
__global__ __launch_bounds__(256) void gemm_wmma_bf16(
    const bf16* __restrict__ A, const bf16* __restrict__ B,
    void* __restrict__ C, const float* __restrict__ mask, float alpha,
    int M, int N, int K, long sA, long sB, long sC) {
  __shared__ bf16 As[2][128][40];   // 32 data + 8 pad bf16 per row
  __shared__ bf16 Bs[2][128][40];   // stored K-major (row = M or N index)

  const int tid  = threadIdx.x;
  const int bm   = blockIdx.y * 128;
  const int bn   = blockIdx.x * 128;
  const bf16* Ab = A + (long)blockIdx.z * sA;
  const bf16* Bb = B + (long)blockIdx.z * sB;

  const int wid  = tid >> 5;          // 0..7 (wave32)
  const int lane = tid & 31;
  const int wm   = wid & 1;           // 2 waves along M (2*64 = 128)
  const int wn   = wid >> 1;          // 4 waves along N (4*32 = 128)
  const int half = lane >> 4;         // lane group (ISA A/B layout)
  const int lrow = lane & 15;

  v8f acc[4][2] = {};

  auto compute_step = [&](const bf16 (*Abuf)[40], const bf16 (*Bbuf)[40]) {
    v16bf af[4], bv[2];
#pragma unroll
    for (int im = 0; im < 4; ++im)
      af[im] = ld_a_frag(&Abuf[wm * 64 + im * 16 + lrow][0], half);
#pragma unroll
    for (int in_ = 0; in_ < 2; ++in_)
      bv[in_] = ld_b_frag(&Bbuf[wn * 32 + in_ * 16 + lrow][0], half);
#pragma unroll
    for (int im = 0; im < 4; ++im)
#pragma unroll
      for (int in_ = 0; in_ < 2; ++in_)
        acc[im][in_] = __builtin_amdgcn_wmma_f32_16x16x32_bf16(
            false, af[im], false, bv[in_], (short)0, acc[im][in_],
            false, false);
  };

  if constexpr (BT) {
    // ---------------- TDM double-buffered NT pipeline ----------------
    const unsigned aAddr[2] = {(unsigned)(unsigned long long)&As[0][0][0],
                               (unsigned)(unsigned long long)&As[1][0][0]};
    const unsigned bAddr[2] = {(unsigned)(unsigned long long)&Bs[0][0][0],
                               (unsigned)(unsigned long long)&Bs[1][0][0]};
    if (wid == 0) {
      tdm_load_tile_bf16(aAddr[0], Ab + (long)bm * K, (unsigned)K,
                         (unsigned)M, (unsigned)K);
      tdm_load_tile_bf16(bAddr[0], Bb + (long)bn * K, (unsigned)K,
                         (unsigned)N, (unsigned)K);
      __builtin_amdgcn_s_wait_tensorcnt(0);
    }
    __syncthreads();
    int buf = 0;
    for (int k0 = 0; k0 < K; k0 += 32) {
      const bool more = (k0 + 32) < K;
      if (more && wid == 0) {   // kick DMA of next tile into the free buffer
        tdm_load_tile_bf16(aAddr[buf ^ 1], Ab + (long)bm * K + (k0 + 32),
                           (unsigned)K, (unsigned)M, (unsigned)K);
        tdm_load_tile_bf16(bAddr[buf ^ 1], Bb + (long)bn * K + (k0 + 32),
                           (unsigned)K, (unsigned)N, (unsigned)K);
      }
      compute_step(As[buf], Bs[buf]);          // overlaps with TDM transfer
      if (more) {
        __syncthreads();                        // all done reading buf
        if (wid == 0) __builtin_amdgcn_s_wait_tensorcnt(0);
        __syncthreads();                        // next buffer ready for all
      }
      buf ^= 1;
    }
  } else {
    // ---------------- manual-staging NN pipeline ----------------
    (void)M;
    for (int k0 = 0; k0 < K; k0 += 32) {
      __syncthreads();
      // A tile (128x32 bf16): 512 x 16B vectors, 2 per thread
#pragma unroll
      for (int it = 0; it < 2; ++it) {
        int vv = tid + it * 256;
        int r = vv >> 2, c8 = (vv & 3) << 3;
        *(uint4*)(&As[0][r][c8]) =
            *(const uint4*)(Ab + (long)(bm + r) * K + k0 + c8);
      }
      // B is [K,N]: vectorized global read along N, transpose into LDS
#pragma unroll
      for (int it = 0; it < 2; ++it) {
        int vv = tid + it * 256;
        int k = vv >> 4, n8 = (vv & 15) << 3;
        uint4 d = *(const uint4*)(Bb + (long)(k0 + k) * N + bn + n8);
        const bf16* e = (const bf16*)&d;
#pragma unroll
        for (int j = 0; j < 8; ++j) Bs[0][n8 + j][k] = e[j];
      }
      if (k0 + 32 < K) {   // global_prefetch_b8 on the next K tile
        __builtin_prefetch(Ab + (long)(bm + (tid >> 1)) * K + k0 + 32, 0, 3);
        __builtin_prefetch(Bb + (long)(k0 + 32 + (tid >> 3)) * N + bn, 0, 3);
      }
      __syncthreads();
      compute_step(As[0], Bs[0]);
    }
  }

  // ---- epilogue (ISA C/D layout: VGPR i -> M = i + 8*half, N = lane&15) ----
  bf16*  C16 = (bf16*)C + (long)blockIdx.z * sC;
  float* C32 = (float*)C + (long)blockIdx.z * sC;
#pragma unroll
  for (int im = 0; im < 4; ++im)
#pragma unroll
    for (int in_ = 0; in_ < 2; ++in_)
#pragma unroll
      for (int i = 0; i < 8; ++i) {
        int m = bm + wm * 64 + im * 16 + half * 8 + i;
        int n = bn + wn * 32 + in_ * 16 + lrow;
        float val = acc[im][in_][i] * alpha;
        if constexpr (ADD_MASK) val += mask[(long)m * N + n];
        if constexpr (OUT_BF16) C16[(long)m * N + n] = f2bf(val);
        else                    C32[(long)m * N + n] = val;
      }
}

// ---------------------------------------------------------------------------
// Row softmax (row length 2048), fp32 in -> bf16 out. 256 threads, wave32.
// ---------------------------------------------------------------------------
__global__ __launch_bounds__(256) void softmax_row_bf16(
    const float* __restrict__ S, bf16* __restrict__ W) {
  const long base = (long)blockIdx.x * 2048;
  const int tid = threadIdx.x;
  const int wid = tid >> 5, lane = tid & 31;
  __shared__ float redm[8];
  __shared__ float reds[8];

  float v[8];
  float m = -3.0e38f;
#pragma unroll
  for (int j = 0; j < 8; ++j) {
    v[j] = S[base + tid + j * 256];
    m = fmaxf(m, v[j]);
  }
#pragma unroll
  for (int off = 16; off > 0; off >>= 1) m = fmaxf(m, __shfl_xor(m, off));
  if (lane == 0) redm[wid] = m;
  __syncthreads();
#pragma unroll
  for (int w = 0; w < 8; ++w) m = fmaxf(m, redm[w]);

  float s = 0.f;
#pragma unroll
  for (int j = 0; j < 8; ++j) {
    v[j] = __expf(v[j] - m);
    s += v[j];
  }
#pragma unroll
  for (int off = 16; off > 0; off >>= 1) s += __shfl_xor(s, off);
  if (lane == 0) reds[wid] = s;
  __syncthreads();
  s = 0.f;
#pragma unroll
  for (int w = 0; w < 8; ++w) s += reds[w];
  const float inv = 1.0f / s;
#pragma unroll
  for (int j = 0; j < 8; ++j) W[base + tid + j * 256] = f2bf(v[j] * inv);
}

// ---------------------------------------------------------------------------
// Launch: x[2,2048,2048], mask[2048,2048], Wq/Wk/Wv/Wo[2048,2048] -> out fp32
// ---------------------------------------------------------------------------
extern "C" void kernel_launch(void* const* d_in, const int* in_sizes, int n_in,
                              void* d_out, int out_size, void* d_ws,
                              size_t ws_size, hipStream_t stream) {
  (void)in_sizes; (void)n_in; (void)out_size; (void)ws_size;

  const float* x    = (const float*)d_in[0];
  const float* mask = (const float*)d_in[1];
  const float* Wq   = (const float*)d_in[2];
  const float* Wk   = (const float*)d_in[3];
  const float* Wv   = (const float*)d_in[4];
  const float* Wo   = (const float*)d_in[5];

  const int BS = 2, S = 2048, D = 2048;
  const size_t XE = (size_t)BS * S * D;   // 8,388,608
  const size_t WE = (size_t)D * D;        // 4,194,304
  const float scale = 0.08838834764831845f;  // 128^-0.5

  // workspace layout (bf16 elements), 128 MiB total
  bf16* xb   = (bf16*)d_ws;
  bf16* Wqb  = xb  + XE;
  bf16* Wkb  = Wqb + WE;
  bf16* Wvb  = Wkb + WE;
  bf16* Wob  = Wvb + WE;
  bf16* qb   = Wob + WE;
  bf16* kb   = qb  + XE;
  bf16* vb   = kb  + XE;
  float* sc  = (float*)(vb + XE);         // scores fp32 [2,2048,2048]
  bf16* wb   = qb;                        // softmax weights reuse q region
  bf16* outb = kb;                        // attention output reuses k region

  dim3 blk(256);

  // 1) convert inputs to bf16
  cvt_f32_to_bf16<<<(int)(XE / 4 / 256), blk, 0, stream>>>(x,  xb,  (int)(XE / 4));
  cvt_f32_to_bf16<<<(int)(WE / 4 / 256), blk, 0, stream>>>(Wq, Wqb, (int)(WE / 4));
  cvt_f32_to_bf16<<<(int)(WE / 4 / 256), blk, 0, stream>>>(Wk, Wkb, (int)(WE / 4));
  cvt_f32_to_bf16<<<(int)(WE / 4 / 256), blk, 0, stream>>>(Wv, Wvb, (int)(WE / 4));
  cvt_f32_to_bf16<<<(int)(WE / 4 / 256), blk, 0, stream>>>(Wo, Wob, (int)(WE / 4));

  // 2) q/k/v = x @ W^T  (NT via TDM, bf16 out)   M=4096, N=2048, K=2048
  dim3 gproj(16, 32, 1);
  gemm_wmma_bf16<true, true, false><<<gproj, blk, 0, stream>>>(
      xb, Wqb, qb, nullptr, 1.0f, 4096, 2048, 2048, 0, 0, 0);
  gemm_wmma_bf16<true, true, false><<<gproj, blk, 0, stream>>>(
      xb, Wkb, kb, nullptr, 1.0f, 4096, 2048, 2048, 0, 0, 0);
  gemm_wmma_bf16<true, true, false><<<gproj, blk, 0, stream>>>(
      xb, Wvb, vb, nullptr, 1.0f, 4096, 2048, 2048, 0, 0, 0);

  // 3) scores = q @ k^T * scale + mask  (NT via TDM, fp32 out, batched)
  dim3 gattn(16, 16, 2);
  gemm_wmma_bf16<true, false, true><<<gattn, blk, 0, stream>>>(
      qb, kb, sc, mask, scale, 2048, 2048, 2048,
      (long)S * D, (long)S * D, (long)S * S);

  // 4) softmax rows -> bf16 weights
  softmax_row_bf16<<<BS * S, blk, 0, stream>>>(sc, wb);

  // 5) out = weights @ v  (NN, manual transpose staging, bf16 out, batched)
  gemm_wmma_bf16<false, true, false><<<gattn, blk, 0, stream>>>(
      wb, vb, outb, nullptr, 1.0f, 2048, 2048, 2048,
      (long)S * S, (long)S * D, (long)S * D);

  // 6) result = out @ Wo^T  (NT via TDM, fp32 out straight to d_out)
  gemm_wmma_bf16<true, false, false><<<gproj, blk, 0, stream>>>(
      outb, Wob, (float*)d_out, nullptr, 1.0f, 4096, 2048, 2048, 0, 0, 0);
}
